// ConvEmbedding_57595511439550
// MI455X (gfx1250) — matrix operand order repaired and checked by reading, not compiled, not run on hardware
//
#include <hip/hip_runtime.h>
#include <math.h>

#define N_BATCH 16384
#define D_DIM   2048
#define E_DIM   512
#define K_DIM   (D_DIM - E_DIM + 1)   // 1537

typedef __attribute__((ext_vector_type(16))) __bf16 v16bf;
typedef __attribute__((ext_vector_type(8)))  float  v8f;

union ABFrag {
    v16bf        v;
    unsigned int u[8];
    uint4        q[2];
};

__device__ __forceinline__ unsigned int f32_to_bf16_rne(float f) {
    unsigned int u = __float_as_uint(f);
    u += 0x7FFFu + ((u >> 16) & 1u);      // round-to-nearest-even
    return u >> 16;
}

// Scatter banded conv weight w (E,K) f32 into dense bf16 Wb (E,D) row-major:
// Wb[e][d] = w[e][d-e] for d in [e, e+K), else 0.  (2 MB in d_ws, L2-resident.)
__global__ void pack_wband(const float* __restrict__ w,
                           unsigned short* __restrict__ wb) {
    const int e = blockIdx.x;
    const float* wrow = w + (size_t)e * K_DIM;
    unsigned short* brow = wb + (size_t)e * D_DIM;
    for (int d = threadIdx.x; d < D_DIM; d += blockDim.x) {
        const int k = d - e;
        const float v = (k >= 0 && k < K_DIM) ? wrow[k] : 0.0f;
        brow[d] = (unsigned short)f32_to_bf16_rne(v);
    }
}

// Fragment-major LDS layout for the bf16 x block: element (row, d) lives at
// byte  (d>>5)*1024 + ((d&31)>>4)*512 + (((d>>3)&1))*256 + row*16 + (d&7)*2
// so lane (m,hh)'s A-fragment halves are at  base + g*1024 (+256), base =
// m*16 + hh*512  ->  1 address add per k-step, conflict-free ds_load_b128.
__global__ __launch_bounds__(256)
void banded_wmma(const float* __restrict__ x,
                 const unsigned short* __restrict__ wb,
                 const float* __restrict__ bias,
                 float* __restrict__ out) {
    __shared__ uint4 xb4[4096];                 // 64 KB bf16 x block

    const int tid  = threadIdx.x;
    const int wave = tid >> 5;                  // 0..7
    const int lane = tid & 31;
    const int n0   = blockIdx.x * 16;

    // ---- Stage: 16x2048 f32 -> bf16 into LDS (one-time conversion). ----
    for (int i = tid; i < 16 * (D_DIM / 2); i += 256) {
        const int row = i & 15;
        const int col = (i >> 4) << 1;          // even element index
        const float2 xv = *(const float2*)(x + (size_t)(n0 + row) * D_DIM + col);
        const unsigned int pk =
            f32_to_bf16_rne(xv.x) | (f32_to_bf16_rne(xv.y) << 16);
        const int kp  = col & 31;
        const int off = ((col >> 5) << 10) + ((kp >> 4) << 9) +
                        (((kp >> 3) & 1) << 8) + (row << 4) + ((kp & 7) << 1);
        *(unsigned int*)((char*)xb4 + off) = pk;
    }
    __syncthreads();

    // WMMA fragment lane decode (ISA 7.12.2):
    const int m  = lane & 15;                   // A row (M)
    const int hh = lane >> 4;                   // lane-half selector
    const int bn = lane & 15;                   // B column (N)
    const int bk = hh << 4;                     // B K offset (0 or 16)

    const char* abase = (const char*)xb4 + (m << 4) + (hh << 9);

    v8f acc_all[4];

    // Wave handles 2 pairs of adjacent e-tiles: eb = (wave + 8*qp)*32,
    // tiles eb and eb+16 share every A fragment (2 WMMAs per A load).
    // Union band window, 64-aligned, is always exactly 1600 cols = 25
    // double-steps (Wb is zero outside the band, so over-reach is free).
    for (int qp = 0; qp < 2; ++qp) {
        const int eb = (wave + (qp << 3)) << 5;
        const int lo = eb & ~63;                // lo+1600 <= 2048 always
        const unsigned short* bcol0 = wb + (size_t)(eb + bn)      * D_DIM + bk;
        const unsigned short* bcol1 = wb + (size_t)(eb + 16 + bn) * D_DIM + bk;

        v8f acc0 = {}, acc1 = {};
        ABFrag aP, aQ, bP0, bP1, bQ0, bQ1;

        {   // prefetch step 0
            const char* ap = abase + ((lo >> 5) << 10);
            aP.q[0]  = *(const uint4*)ap;
            aP.q[1]  = *(const uint4*)(ap + 256);
            bP0.q[0] = *(const uint4*)(bcol0 + lo);
            bP0.q[1] = *(const uint4*)(bcol0 + lo + 8);
            bP1.q[0] = *(const uint4*)(bcol1 + lo);
            bP1.q[1] = *(const uint4*)(bcol1 + lo + 8);
        }

        for (int it = 0; it < 25; ++it) {
            const int kk = lo + (it << 6);

            // prefetch odd step (kk+32) while WMMAs consume even step
            const char* apq = abase + (((kk + 32) >> 5) << 10);
            aQ.q[0]  = *(const uint4*)apq;
            aQ.q[1]  = *(const uint4*)(apq + 256);
            bQ0.q[0] = *(const uint4*)(bcol0 + kk + 32);
            bQ0.q[1] = *(const uint4*)(bcol0 + kk + 40);
            bQ1.q[0] = *(const uint4*)(bcol1 + kk + 32);
            bQ1.q[1] = *(const uint4*)(bcol1 + kk + 40);

            acc0 = __builtin_amdgcn_wmma_f32_16x16x32_bf16(
                false, aP.v, false, bP0.v, (short)0, acc0, false, false);
            acc1 = __builtin_amdgcn_wmma_f32_16x16x32_bf16(
                false, aP.v, false, bP1.v, (short)0, acc1, false, false);

            // prefetch next even step (redundant reload of `lo` on last trip)
            const int kn = (it < 24) ? (kk + 64) : lo;
            const char* apn = abase + ((kn >> 5) << 10);
            aP.q[0]  = *(const uint4*)apn;
            aP.q[1]  = *(const uint4*)(apn + 256);
            bP0.q[0] = *(const uint4*)(bcol0 + kn);
            bP0.q[1] = *(const uint4*)(bcol0 + kn + 8);
            bP1.q[0] = *(const uint4*)(bcol1 + kn);
            bP1.q[1] = *(const uint4*)(bcol1 + kn + 8);

            acc0 = __builtin_amdgcn_wmma_f32_16x16x32_bf16(
                false, aQ.v, false, bQ0.v, (short)0, acc0, false, false);
            acc1 = __builtin_amdgcn_wmma_f32_16x16x32_bf16(
                false, aQ.v, false, bQ1.v, (short)0, acc1, false, false);
        }

        const float bv0 = bias[eb + bn];
        const float bv1 = bias[eb + 16 + bn];
#pragma unroll
        for (int r = 0; r < 8; ++r) { acc0[r] += bv0; acc1[r] += bv1; }
        acc_all[(qp << 1) + 0] = acc0;
        acc_all[(qp << 1) + 1] = acc1;
    }

    // ---- Fused F.normalize(dim=1), output stays in registers. ----
    __syncthreads();                            // all waves done reading xb
    float* rowssq = (float*)xb4;                // reuse 64 B of LDS
    if (tid < 16) rowssq[tid] = 0.0f;
    __syncthreads();

    float part[8];
#pragma unroll
    for (int r = 0; r < 8; ++r) {               // row = r + hh*8
        float s = 0.0f;
#pragma unroll
        for (int et = 0; et < 4; ++et) s += acc_all[et][r] * acc_all[et][r];
        s += __shfl_xor(s, 1, 32);              // reduce 16-lane hh-group
        s += __shfl_xor(s, 2, 32);
        s += __shfl_xor(s, 4, 32);
        s += __shfl_xor(s, 8, 32);
        part[r] = s;
    }
    if (bn == 0) {                              // one lane per hh-group
#pragma unroll
        for (int r = 0; r < 8; ++r)
            atomicAdd(&rowssq[r + (hh << 3)], part[r]);
    }
    __syncthreads();

#pragma unroll
    for (int r = 0; r < 8; ++r) {
        const int row = r + (hh << 3);
        const float scale = 1.0f / fmaxf(sqrtf(rowssq[row]), 1e-12f);
        float* orow = out + (size_t)(n0 + row) * E_DIM + bn;
#pragma unroll
        for (int et = 0; et < 4; ++et) {
            const int e0 = ((wave + ((et >> 1) << 3)) << 5) + ((et & 1) << 4);
            orow[e0] = acc_all[et][r] * scale;
        }
    }
}

extern "C" void kernel_launch(void* const* d_in, const int* in_sizes, int n_in,
                              void* d_out, int out_size, void* d_ws, size_t ws_size,
                              hipStream_t stream) {
    const float* x = (const float*)d_in[0];      // (N, D) f32
    const float* w = (const float*)d_in[1];      // (E, K) f32
    const float* b = (const float*)d_in[2];      // (E,)   f32
    float* out = (float*)d_out;                  // (N, E) f32
    unsigned short* wb = (unsigned short*)d_ws;  // E*D bf16 = 2 MB scratch

    pack_wband<<<E_DIM, 256, 0, stream>>>(w, wb);
    banded_wmma<<<N_BATCH / 16, 256, 0, stream>>>(x, wb, b, out);
}
